// LMUFFT_32512902431367
// MI455X (gfx1250) — compile-verified, matrix operand order checked by reading, not run on hardware
//
#include <hip/hip_runtime.h>

// ---------------------------------------------------------------------------
// LMU-FFT forward on MI455X (gfx1250, wave32, WMMA, TDM).
// Causal convolution as Toeplitz GEMM with f16 WMMA / f32 acc:
//   m[o,t] = sum_tau H[o,tau] * u[t-tau]   ->  A = H tile, B = sliding window
// Concat buffer [B,L,1280] feeds a second f16 WMMA GEMM for h.
// A-operand tiles are staged into LDS by the Tensor Data Mover with a
// double-buffered (TENSORcnt<=1) pipeline; TDM LDS-padding gives 80B rows.
// ---------------------------------------------------------------------------

typedef __attribute__((ext_vector_type(16))) _Float16 v16h;
typedef __attribute__((ext_vector_type(8)))  float    v8f;

typedef __attribute__((ext_vector_type(4))) unsigned int tdm_v4u;
typedef __attribute__((ext_vector_type(8))) int          tdm_v8i;
typedef __attribute__((ext_vector_type(4))) int          tdm_v4i;

#define LMU_B   8
#define LMU_L   4096
#define LMU_IN  256
#define LMU_MD  4
#define LMU_ORD 256
#define LMU_HID 512
#define MSZ     1280      // MD*ORD + IN
#define UPAD    128       // zero padding in front/back of each u row
#define USTRIDE (UPAD + LMU_L + UPAD)   // 4352 halves per (b,d) row

#if defined(__gfx1250__) && __has_builtin(__builtin_amdgcn_tensor_load_to_lds)
#define HAS_TDM 1
#else
#define HAS_TDM 0
#endif

union QH { uint4 q[2]; v16h h; };   // 32B fragment: two b128 LDS loads
union Q8 { _Float16 h[8]; uint4 q; };

// ---------------------------------------------------------------------------
// Tensor Data Mover: async 2D f16 tile load Global->LDS.
// Writes tile rows contiguously, inserting 16B of LDS padding after every
// 64B of data (pad_interval = 16 DWORDs, pad_amount = 4 DWORDs) => 80B rows.
// Descriptor fields per CDNA5 ISA 8.3/8.4 (D# group0/group1).
// ---------------------------------------------------------------------------
__device__ __forceinline__ unsigned lds_off(const void* p) {
    return (unsigned)(size_t)p;     // low 32 bits of generic addr = LDS byte offset
}

__device__ __forceinline__ void tdm_load_tile_f16(
    const void* gaddr, unsigned lds_byte_off,
    unsigned tensor_w, unsigned tensor_h,      // tensor dims (elements)
    unsigned tile_w, unsigned tile_h,          // tile dims (elements)
    unsigned row_stride_elems) {
#if HAS_TDM
    unsigned long long ga = (unsigned long long)(size_t)gaddr;
    tdm_v4u g0;
    g0[0] = 1u;                                        // count=1, user D#
    g0[1] = lds_byte_off;                              // lds_addr
    g0[2] = (unsigned)(ga & 0xffffffffu);              // global_addr[31:0]
    g0[3] = (unsigned)((ga >> 32) & 0x1ffffffu)        // global_addr[56:32]
          | (2u << 30);                                // type=2 ("image")
    tdm_v8i g1;
    g1[0] = (1 << 16)                                  // data_size = 2 bytes
          | (1 << 20)                                  // pad_enable
          | (3 << 22)                                  // pad_interval: 16 DWORDs
          | (3 << 25);                                 // pad_amount: 4 DWORDs
    g1[1] = (int)((tensor_w & 0xffffu) << 16);         // tensor_dim0[15:0]
    g1[2] = (int)((tensor_w >> 16) | ((tensor_h & 0xffffu) << 16));
    g1[3] = (int)((tensor_h >> 16) | (tile_w << 16));  // tile_dim0
    g1[4] = (int)(tile_h & 0xffffu);                   // tile_dim1 (tile_dim2=0)
    g1[5] = (int)row_stride_elems;                     // tensor_dim0_stride[31:0]
    g1[6] = 0;                                         // stride hi / dim1_stride lo
    g1[7] = 0;
    tdm_v4i z4 = {0, 0, 0, 0};
#if __clang_major__ >= 23
    tdm_v8i z8 = {0, 0, 0, 0, 0, 0, 0, 0};
    __builtin_amdgcn_tensor_load_to_lds(g0, g1, z4, z4, z8, 0);
#else
    __builtin_amdgcn_tensor_load_to_lds(g0, g1, z4, z4, 0);
#endif
#else
    (void)gaddr; (void)lds_byte_off; (void)tensor_w; (void)tensor_h;
    (void)tile_w; (void)tile_h; (void)row_stride_elems;
#endif
}

// Wait until at most N tensor ops in flight (immediate operand).
__device__ __forceinline__ void tdm_wait_le(bool one_in_flight) {
#if HAS_TDM
    if (one_in_flight) __builtin_amdgcn_s_wait_tensorcnt(1);
    else               __builtin_amdgcn_s_wait_tensorcnt(0);
#else
    (void)one_in_flight;
#endif
}

// ---------------------------------------------------------------------------
// Small prep kernels
// ---------------------------------------------------------------------------
__global__ void cvt_f16_kernel(const float* __restrict__ s, _Float16* __restrict__ d, int n) {
    int i = blockIdx.x * blockDim.x + threadIdx.x;
    if (i < n) d[i] = (_Float16)s[i];
}

__global__ void zero_f16_kernel(_Float16* __restrict__ d, int n) {
    int i = blockIdx.x * blockDim.x + threadIdx.x;
    if (i < n) d[i] = (_Float16)0.f;
}

// ---------------------------------------------------------------------------
// Stage 1: u = relu(x @ Wu^T + b)  (67 MFLOP, memory bound)
// One block per (b,l) row. Also drops x as f16 into the concat buffer tail.
// ---------------------------------------------------------------------------
__global__ __launch_bounds__(256)
void u_proj_kernel(const float* __restrict__ x, const float* __restrict__ Wu,
                   const float* __restrict__ Wub,
                   _Float16* __restrict__ uo, _Float16* __restrict__ ih) {
    __shared__ float xs[LMU_IN];
    const int row = blockIdx.x;          // b*L + l
    const int tid = threadIdx.x;

    const float xv = x[(size_t)row * LMU_IN + tid];
    xs[tid] = xv;
    // x occupies columns [1024,1280) of the concat buffer
    ih[(size_t)row * MSZ + LMU_MD * LMU_ORD + tid] = (_Float16)xv;
    __syncthreads();

    const int wave = tid >> 5;
    const int lane = tid & 31;
    if (wave < LMU_MD) {                 // one wave32 per memory channel
        const int dch = wave;
        float p = 0.f;
        #pragma unroll
        for (int j = 0; j < 8; ++j) {
            const int i = lane + j * 32;
            p += xs[i] * Wu[dch * LMU_IN + i];
        }
        #pragma unroll
        for (int off = 16; off > 0; off >>= 1) p += __shfl_xor(p, off, 32);
        if (lane == 0) {
            float v = p + Wub[dch];
            v = v > 0.f ? v : 0.f;
            const int b = row >> 12, l = row & (LMU_L - 1);
            uo[(size_t)(b * LMU_MD + dch) * USTRIDE + UPAD + l] = (_Float16)v;
        }
    }
}

// ---------------------------------------------------------------------------
// Stage 2: Toeplitz GEMM conv via WMMA.
// Block: 256 threads (8 waves). Output tile: 256(o) x 64(t) per block.
// Wave w owns o in [32w, 32w+32): 2 o-tiles x 4 t-tiles of 16x16 f32 acc.
// K loop over tau in steps of 32 (causal: tau0 <= t0+63).
// H slab double-buffered via TDM; Toeplitz B tiles built from a u window.
// ---------------------------------------------------------------------------
__global__ __launch_bounds__(256)
void conv_wmma_kernel(const _Float16* __restrict__ u, const _Float16* __restrict__ H,
                      _Float16* __restrict__ ih) {
    // Hs rows: 64B data + 16B pad (TDM pad / manual) -> 80B stride.
    __shared__ uint4     Hs[2][256 * 5];      // 2 x 20 KB
    __shared__ _Float16  us[2][160];          // sliding windows of u
    __shared__ uint4     Bs[2][4 * 32 * 2];   // materialized Toeplitz B tiles

    const int tid  = threadIdx.x;
    const int lane = tid & 31;
    const int wave = tid >> 5;
    const int hlf  = lane >> 4;               // lane half (K-half selector)
    const int ln16 = lane & 15;

    const int t0 = blockIdx.x * 64;
    const int bd = blockIdx.y;                // b*MD + d
    const int b  = bd >> 2;
    const int d  = bd & 3;

    const _Float16* urow = u + (size_t)bd * USTRIDE + UPAD;
    const int nsteps = (t0 + 64) / 32;        // causal K extent
    const unsigned hs_base[2] = { lds_off(&Hs[0][0]), lds_off(&Hs[1][0]) };

    auto stage_H = [&](int step, int buf) {
#if HAS_TDM
        if (wave == 0)
            tdm_load_tile_f16(H + step * 32, hs_base[buf],
                              LMU_L, LMU_ORD,   // tensor 4096 x 256
                              32, LMU_ORD,      // tile   32 x 256
                              LMU_L);           // row stride 4096 elems
#else
        const uint4* src = (const uint4*)(H + (size_t)tid * LMU_L + step * 32);
        Hs[buf][tid * 5 + 0] = src[0];
        Hs[buf][tid * 5 + 1] = src[1];
        Hs[buf][tid * 5 + 2] = src[2];
        Hs[buf][tid * 5 + 3] = src[3];
        if (step * 32 + 64 <= LMU_L)
            __builtin_prefetch(src + 4, 0, 1);  // global_prefetch_b8
#endif
    };
    auto stage_u = [&](int step, int buf) {
        if (tid < 128) us[buf][tid] = urow[t0 - step * 32 - 32 + tid];
    };

    v8f acc[2][4];
    {
        v8f z = {};
        #pragma unroll
        for (int i = 0; i < 2; ++i)
            #pragma unroll
            for (int j = 0; j < 4; ++j) acc[i][j] = z;
    }

    // prologue: prime the pipeline
    stage_H(0, 0);
    stage_u(0, 0);
    __syncthreads();

    for (int s = 0; s < nsteps; ++s) {
        const int cur = s & 1, nxt = cur ^ 1;
        const bool more = (s + 1) < nsteps;

        // 1. materialize the 4 Toeplitz B tiles for this step (all waves share)
        {
            const int combo = tid >> 1;       // (nt, fragment-lane)
            const int part  = tid & 1;        // which 8-element half of v16h
            const int bl    = combo & 31;
            const int nt    = combo >> 5;
            const int n     = nt * 16 + (bl & 15);
            const int bh    = bl >> 4;
            const int kbase = part * 16 + bh * 8;
            Q8 pk;
            #pragma unroll
            for (int e = 0; e < 8; ++e)
                pk.h[e] = us[cur][n + 32 - (kbase + e)];   // B[k][n] = u[t-tau]
            Bs[cur][combo * 2 + part] = pk.q;
        }
        // 2. issue next slab's DMA while this step computes
        if (more) { stage_H(s + 1, nxt); stage_u(s + 1, nxt); }
        // 3. ensure the CURRENT slab has landed (<=1 older op outstanding)
        tdm_wait_le(more);
        __syncthreads();

        // 4. fragments + 8 WMMAs
        QH afr[2];
        #pragma unroll
        for (int ot = 0; ot < 2; ++ot) {
            const int row = wave * 32 + ot * 16 + ln16;
            afr[ot].q[0] = Hs[cur][row * 5 + hlf];         // k = hlf*8 .. +7
            afr[ot].q[1] = Hs[cur][row * 5 + 2 + hlf];     // k = 16+hlf*8 .. +7
        }
        #pragma unroll
        for (int tt = 0; tt < 4; ++tt) {
            QH bfr;
            bfr.q[0] = Bs[cur][(tt * 32 + lane) * 2 + 0];
            bfr.q[1] = Bs[cur][(tt * 32 + lane) * 2 + 1];
            #pragma unroll
            for (int ot = 0; ot < 2; ++ot)
                acc[ot][tt] = __builtin_amdgcn_wmma_f32_16x16x32_f16(
                    false, afr[ot].h, false, bfr.h, (short)0, acc[ot][tt],
                    false, false);
        }
        __syncthreads();   // release Hs[cur] for the DMA issued 2 steps ahead
    }

    // --- epilogue: packed b128 stores into concat buffer [B,L,1280]
    // Accumulator rows r map to consecutive channels o = obase + r.
    _Float16* outp = ih + (size_t)b * LMU_L * MSZ + d * LMU_ORD;
    #pragma unroll
    for (int ot = 0; ot < 2; ++ot) {
        const int obase = wave * 32 + ot * 16 + hlf * 8;   // multiple of 8
        #pragma unroll
        for (int tt = 0; tt < 4; ++tt) {
            const int t = t0 + tt * 16 + ln16;
            Q8 pk;
            #pragma unroll
            for (int r = 0; r < 8; ++r) pk.h[r] = (_Float16)acc[ot][tt][r];
            *(uint4*)(outp + (size_t)t * MSZ + obase) = pk.q;   // 16B aligned
        }
    }
}

// ---------------------------------------------------------------------------
// Stage 3: h = relu(concat @ Wh^T + b).  [32768 x 1280] x [1280 x 512].
// Block tile 128(M) x 128(N); wave grid 2x4, each wave 4x2 acc tiles.
// A slab double-buffered via TDM; B slab swizzled to fragment layout.
// ---------------------------------------------------------------------------
__global__ __launch_bounds__(256)
void hgemm_out_kernel(const _Float16* __restrict__ ih, const _Float16* __restrict__ Wh,
                      const float* __restrict__ bias, float* __restrict__ out) {
    __shared__ uint4 As[2][128 * 5];     // 2 x 10 KB, 80B row stride
    __shared__ uint4 Bs[2][8 * 32 * 2];  // 2 x 8 KB, fragment layout

    const int tid  = threadIdx.x;
    const int lane = tid & 31;
    const int wave = tid >> 5;
    const int hlf  = lane >> 4;
    const int ln16 = lane & 15;

    const int row0 = blockIdx.x * 128;
    const int col0 = blockIdx.y * 128;
    const int wm = wave >> 2;         // 0..1 -> 64 rows each
    const int wn = wave & 3;          // 0..3 -> 32 cols each
    const unsigned as_base[2] = { lds_off(&As[0][0]), lds_off(&As[1][0]) };
    const int nsteps = MSZ / 32;      // 40

    auto stage_A = [&](int step, int buf) {
#if HAS_TDM
        if (wave == 0)
            tdm_load_tile_f16(ih + (size_t)row0 * MSZ + step * 32, as_base[buf],
                              MSZ, LMU_B * LMU_L,   // tensor 1280 x 32768
                              32, 128,              // tile   32 x 128
                              MSZ);                 // row stride 1280 elems
#else
        const int row = tid >> 1, part = tid & 1;
        const uint4* src = (const uint4*)(ih + (size_t)(row0 + row) * MSZ + step * 32) + part * 2;
        As[buf][row * 5 + 2 * part + 0] = src[0];
        As[buf][row * 5 + 2 * part + 1] = src[1];
#endif
    };

    v8f acc[4][2];
    {
        v8f z = {};
        #pragma unroll
        for (int i = 0; i < 4; ++i)
            #pragma unroll
            for (int j = 0; j < 2; ++j) acc[i][j] = z;
    }

    stage_A(0, 0);   // prime pipeline

    for (int s = 0; s < nsteps; ++s) {
        const int cur = s & 1, nxt = cur ^ 1;
        const bool more = (s + 1) < nsteps;
        const int k0 = s * 32;

        // 1. stage B: Wh rows -> fragment layout (lane-contiguous 32B)
        {
            const int n = tid >> 1, h = tid & 1;
            const _Float16* wr = Wh + (size_t)(col0 + n) * MSZ + k0;
            uint4 s1 = *(const uint4*)(wr + h * 8);         // k = h*8 .. h*8+7
            uint4 s2 = *(const uint4*)(wr + 16 + h * 8);    // k = 16+h*8 ..
            const int combo = (n >> 4) * 32 + (h << 4) + (n & 15);
            Bs[cur][combo * 2 + 0] = s1;
            Bs[cur][combo * 2 + 1] = s2;
            if (more) __builtin_prefetch(wr + 32, 0, 1);    // global_prefetch_b8
        }
        // 2. issue next A slab, 3. wait for current one
        if (more) stage_A(s + 1, nxt);
        tdm_wait_le(more);
        __syncthreads();

        // 4. fragments + 8 WMMAs
        QH afr[4];
        #pragma unroll
        for (int mt = 0; mt < 4; ++mt) {
            const int r = wm * 64 + mt * 16 + ln16;
            afr[mt].q[0] = As[cur][r * 5 + hlf];
            afr[mt].q[1] = As[cur][r * 5 + 2 + hlf];
        }
        #pragma unroll
        for (int nt = 0; nt < 2; ++nt) {
            QH bfr;
            const int combo = (wn * 2 + nt) * 32 + lane;
            bfr.q[0] = Bs[cur][combo * 2 + 0];
            bfr.q[1] = Bs[cur][combo * 2 + 1];
            #pragma unroll
            for (int mt = 0; mt < 4; ++mt)
                acc[mt][nt] = __builtin_amdgcn_wmma_f32_16x16x32_f16(
                    false, afr[mt].h, false, bfr.h, (short)0, acc[mt][nt],
                    false, false);
        }
        __syncthreads();   // release As[cur]/Bs[cur]
    }

    // --- epilogue: bias + relu, f32 stores
    #pragma unroll
    for (int nt = 0; nt < 2; ++nt) {
        const int col = col0 + wn * 32 + nt * 16 + ln16;
        const float bv = bias[col];
        #pragma unroll
        for (int mt = 0; mt < 4; ++mt) {
            const int rbase = row0 + wm * 64 + mt * 16 + hlf * 8;
            #pragma unroll
            for (int r = 0; r < 8; ++r) {
                float v = acc[mt][nt][r] + bv;
                out[(size_t)(rbase + r) * LMU_HID + col] = v > 0.f ? v : 0.f;
            }
        }
    }
}

// ---------------------------------------------------------------------------
// h[:, -1, :] -> tail of d_out
// ---------------------------------------------------------------------------
__global__ void last_row_kernel(const float* __restrict__ h, float* __restrict__ dst) {
    int i = blockIdx.x * blockDim.x + threadIdx.x;
    if (i < LMU_B * LMU_HID) {
        int b = i >> 9, j = i & (LMU_HID - 1);
        dst[i] = h[((size_t)b * LMU_L + LMU_L - 1) * LMU_HID + j];
    }
}

// ---------------------------------------------------------------------------
extern "C" void kernel_launch(void* const* d_in, const int* in_sizes, int n_in,
                              void* d_out, int out_size, void* d_ws, size_t ws_size,
                              hipStream_t stream) {
    (void)in_sizes; (void)n_in; (void)out_size; (void)ws_size;
    const float* x    = (const float*)d_in[0];
    const float* Wu_w = (const float*)d_in[1];
    const float* Wu_b = (const float*)d_in[2];
    const float* Wh_w = (const float*)d_in[3];
    const float* Wh_b = (const float*)d_in[4];
    const float* H    = (const float*)d_in[5];
    float* out = (float*)d_out;

    char* ws = (char*)d_ws;
    size_t off = 0;
    auto alloc = [&](size_t bytes) {
        void* p = ws + off;
        off = (off + bytes + 255) & ~(size_t)255;
        return p;
    };
    _Float16* u16  = (_Float16*)alloc((size_t)LMU_B * LMU_MD * USTRIDE * 2);   // 272 KB
    _Float16* H16  = (_Float16*)alloc((size_t)LMU_ORD * LMU_L * 2);            // 2 MB
    _Float16* Wh16 = (_Float16*)alloc((size_t)LMU_HID * MSZ * 2);              // 1.3 MB
    _Float16* ih   = (_Float16*)alloc((size_t)LMU_B * LMU_L * MSZ * 2);        // 84 MB

    const int nU  = LMU_B * LMU_MD * USTRIDE;
    const int nH  = LMU_ORD * LMU_L;
    const int nWh = LMU_HID * MSZ;

    zero_f16_kernel<<<(nU + 255) / 256, 256, 0, stream>>>(u16, nU);
    cvt_f16_kernel<<<(nH + 255) / 256, 256, 0, stream>>>(H, H16, nH);
    cvt_f16_kernel<<<(nWh + 255) / 256, 256, 0, stream>>>(Wh_w, Wh16, nWh);

    u_proj_kernel<<<LMU_B * LMU_L, 256, 0, stream>>>(x, Wu_w, Wu_b, u16, ih);

    conv_wmma_kernel<<<dim3(LMU_L / 64, LMU_B * LMU_MD), 256, 0, stream>>>(u16, H16, ih);

    hgemm_out_kernel<<<dim3(LMU_B * LMU_L / 128, LMU_HID / 128), 256, 0, stream>>>(
        ih, Wh16, Wh_b, out);

    last_row_kernel<<<(LMU_B * LMU_HID + 255) / 256, 256, 0, stream>>>(
        out, out + (size_t)LMU_B * LMU_L * LMU_HID);
}
